// DelayedHORN_71889162600899
// MI455X (gfx1250) — matrix operand order repaired and checked by reading, not compiled, not run on hardware
//
#include <hip/hip_runtime.h>

// ---- explicit ext-vector types ----
typedef __attribute__((ext_vector_type(16))) __bf16 v16bf;
typedef __attribute__((ext_vector_type(8)))  float  v8f;
typedef __attribute__((ext_vector_type(4)))  int    i32x4;
typedef __attribute__((ext_vector_type(4)))  float  f32x4;

#define T_LEN 784
#define B_TOT 512
#define N_DIM 128
#define C_DIM 10
#define DELAYN 5
#define BT 16      // batch rows per workgroup (WMMA M)
#define TCH 112    // timesteps staged per LDS chunk (784 = 7*112)

struct Frag32B { i32x4 q0, q1; };   // 32 bytes == v16bf

__device__ __forceinline__ float fast_tanh(float a) {
#if __has_builtin(__builtin_amdgcn_tanhf)
    return __builtin_amdgcn_tanhf(a);            // single v_tanh_f32 (TRANS)
#else
    float e = __builtin_amdgcn_exp2f(a * 2.8853900817779268f);  // exp(2a)
    return 1.0f - 2.0f * __builtin_amdgcn_rcpf(e + 1.0f);
#endif
}

__global__ __launch_bounds__(256)
void horn_step_kernel(const float* __restrict__ in_seq,   // [512,784]
                      const float* __restrict__ w_ih,     // [128]
                      const float* __restrict__ b_ih,     // [128]
                      const float* __restrict__ w_hh,     // [128,128]
                      const float* __restrict__ b_hh,     // [128]
                      const float* __restrict__ w_ro,     // [10,128]
                      const float* __restrict__ b_ro,     // [10]
                      float* __restrict__ out)            // [512,10]
{
    __shared__ __align__(16) __bf16 whh_lds[N_DIM * N_DIM];        // 32 KB
    __shared__ __align__(16) __bf16 ybuf[DELAYN * BT * N_DIM];     // 20 KB (y-history ring)
    __shared__ __align__(16) float  s_tile[TCH * BT];              // 7 KB  (input chunk, transposed)

    const int tid   = threadIdx.x;
    const int wv    = tid >> 5;          // wave id 0..7 -> N-chunk
    const int lane  = tid & 31;
    const int hi    = lane >> 4;         // half-wave (K split for A/B frags)
    const int lm    = lane & 15;
    const int btile = blockIdx.x * BT;
    const int n     = wv * 16 + lm;      // output unit owned by this lane (C/D column)

    // ---- stage w_hh as bf16 into LDS ----
    for (int i = tid; i < N_DIM * N_DIM; i += 256)
        whh_lds[i] = (__bf16)w_hh[i];
    // ---- zero y-history ring ----
    int* ybuf_i = (int*)ybuf;
    for (int i = tid; i < (DELAYN * BT * N_DIM) / 2; i += 256)
        ybuf_i[i] = 0;
    __syncthreads();

    // ---- hoist loop-invariant B fragments: B[k][n] = w_hh[n][k] ----
    // 16-bit B 32x16 layout: lanes 0-15 hold K={0..7,16..23} (N=lane), lanes 16-31 hold K={8..15,24..31}.
    v16bf bfr[4];
    #pragma unroll
    for (int kc = 0; kc < 4; ++kc) {
        const __bf16* p = &whh_lds[n * N_DIM + kc * 32 + hi * 8];
        Frag32B f;
        f.q0 = *(const i32x4*)(p);
        f.q1 = *(const i32x4*)(p + 16);
        bfr[kc] = __builtin_bit_cast(v16bf, f);
    }

    const float wih = w_ih[n];
    const float bih = b_ih[n];
    const float bhh = b_hh[n];
    const float inv_sqrt_n = 0.08838834764831845f;   // 1/sqrt(128)

    // persistent oscillator state in WMMA C/D register layout:
    // element r of this lane <-> (b = btile + r + 8*hi, unit n)
    float x[8], y[8];
    #pragma unroll
    for (int r = 0; r < 8; ++r) { x[r] = 0.0f; y[r] = 0.0f; }

    int slot = 0;
    for (int t0 = 0; t0 < T_LEN; t0 += TCH) {
        // ---- stage input chunk transposed: s_tile[j][i] = in[btile+i][t0+j] ----
        __syncthreads();
        for (int idx = tid; idx < TCH * BT; idx += 256) {
            int i = idx / TCH, j = idx % TCH;  // adjacent threads read contiguous t
            s_tile[j * BT + i] = in_seq[(size_t)(btile + i) * T_LEN + t0 + j];
        }
        if (t0 + TCH < T_LEN)   // touch next chunk early (global_prefetch_b8)
            __builtin_prefetch(&in_seq[(size_t)(btile + lm) * T_LEN + t0 + TCH + hi * 56], 0, 1);
        __syncthreads();

        for (int tl = 0; tl < TCH; ++tl) {
            // ---- I_rec = y(t-5) @ w_hh^T + b_hh : 4 chained bf16 WMMAs ----
            v8f acc;
            #pragma unroll
            for (int r = 0; r < 8; ++r) acc[r] = bhh;
            #pragma unroll
            for (int kc = 0; kc < 4; ++kc) {
                // 16-bit A 16x32 layout: lanes 0-15 K={0..7,16..23}, lanes 16-31 K={8..15,24..31}
                const __bf16* p = &ybuf[slot * (BT * N_DIM) + lm * N_DIM + kc * 32 + hi * 8];
                Frag32B f;
                f.q0 = *(const i32x4*)(p);
                f.q1 = *(const i32x4*)(p + 16);
                v16bf a = __builtin_bit_cast(v16bf, f);
                acc = __builtin_amdgcn_wmma_f32_16x16x32_bf16(
                          false, a, false, bfr[kc], (short)0, acc, false, false);
            }
            const f32x4 s0 = *(const f32x4*)&s_tile[tl * BT + hi * 8];
            const f32x4 s1 = *(const f32x4*)&s_tile[tl * BT + hi * 8 + 4];

            // ---- split barrier: announce "done reading ring slot" early.
            //      s_wait_dscnt 0 guarantees our LDS reads have completed in LDS.
            asm volatile("s_wait_dscnt 0\n\ts_barrier_signal -1" ::: "memory");

            #pragma unroll
            for (int r = 0; r < 8; ++r) {
                float sv = (r < 4) ? s0[r & 3] : s1[r & 3];
                float A  = (sv * wih + bih + acc[r]) * inv_sqrt_n;
                float accel = fast_tanh(A) - 0.2f * y[r] - x[r];
                x[r] = x[r] + y[r];          // x_new = x + H*y (old y)
                y[r] = y[r] + accel;         // y_new = y + H*accel
            }

            // wait until *all* waves finished reading before overwriting the slot.
            // The dummy "v" operands pin the state update BEFORE the wait, so the
            // barrier round-trip is hidden behind the tanh/oscillator math.
            asm volatile("s_barrier_wait -1"
                         :
                         : "v"(y[0]), "v"(y[1]), "v"(y[2]), "v"(y[3]),
                           "v"(y[4]), "v"(y[5]), "v"(y[6]), "v"(y[7])
                         : "memory");
            #pragma unroll
            for (int r = 0; r < 8; ++r)
                ybuf[slot * (BT * N_DIM) + (r + 8 * hi) * N_DIM + n] = (__bf16)y[r];

            slot = (slot == DELAYN - 1) ? 0 : slot + 1;
        }
    }

    // ---- readout: out[b][c] = x_final[b] . w_ro[c] + b_ro[c] ----
    __syncthreads();
    float* xout = (float*)ybuf;              // reuse ring buffer (8 KB needed)
    #pragma unroll
    for (int r = 0; r < 8; ++r)
        xout[(r + 8 * hi) * N_DIM + n] = x[r];
    __syncthreads();
    for (int idx = tid; idx < BT * C_DIM; idx += 256) {
        int b = idx / C_DIM, c = idx % C_DIM;
        float sum = b_ro[c];
        #pragma unroll 8
        for (int k = 0; k < N_DIM; ++k)
            sum += xout[b * N_DIM + k] * w_ro[c * N_DIM + k];
        out[(size_t)(btile + b) * C_DIM + c] = sum;
    }
}

extern "C" void kernel_launch(void* const* d_in, const int* in_sizes, int n_in,
                              void* d_out, int out_size, void* d_ws, size_t ws_size,
                              hipStream_t stream) {
    (void)in_sizes; (void)n_in; (void)out_size; (void)d_ws; (void)ws_size;
    const float* in_seq = (const float*)d_in[0];
    const float* w_ih   = (const float*)d_in[1];
    const float* b_ih   = (const float*)d_in[2];
    const float* w_hh   = (const float*)d_in[3];
    const float* b_hh   = (const float*)d_in[4];
    const float* w_ro   = (const float*)d_in[5];
    const float* b_ro   = (const float*)d_in[6];
    float* out = (float*)d_out;

    horn_step_kernel<<<dim3(B_TOT / BT), dim3(256), 0, stream>>>(
        in_seq, w_ih, b_ih, w_hh, b_hh, w_ro, b_ro, out);
}
